// SelfAttention2d_75187697484072
// MI455X (gfx1250) — compile-verified, matrix operand order checked by reading, not run on hardware
//
#include <hip/hip_runtime.h>
#include <hip/hip_bf16.h>
#include <stdint.h>

// ---------------------------------------------------------------------------
// SAGAN self-attention on MI455X (gfx1250), bf16 WMMA + register-resident
// flash attention. N=16, C=512, H=W=64 (HW=4096), C_BAR=64, C_HAT=256.
// ---------------------------------------------------------------------------

typedef __attribute__((ext_vector_type(16))) __bf16 bf16x16;  // A/B WMMA frag
typedef __attribute__((ext_vector_type(8)))  __bf16 bf16x8;
typedef __attribute__((ext_vector_type(8)))  float  f32x8;    // C/D WMMA frag

__device__ __forceinline__ __bf16 f2bf(float f) {
  return (__bf16)f;   // native fptrunc -> hardware bf16 cvt (RNE)
}

// Load 16(row) x 32(K) bf16 fragment.  Per CDNA5 ISA layout:
//   lane -> row (M for A, N for B), halves h -> K = h + 8*(h>=8) + 8*(lane>=16)
// src is row-major [row][ld] with the K-window starting at src.
// Union keeps the two 16B loads contiguous so they land directly in the
// 8-VGPR fragment without cross-register moves.
__device__ __forceinline__ bf16x16 load_frag(const __bf16* src, int ld, int lane) {
  const __bf16* p = src + (size_t)(lane & 15) * ld + ((lane >> 4) << 3);
  union { bf16x16 v; bf16x8 h[2]; } u;
  u.h[0] = *(const bf16x8*)(p);        // K = 0..7   (+8 upper half-wave)
  u.h[1] = *(const bf16x8*)(p + 16);   // K = 16..23 (+8 upper half-wave)
  return u.v;
}

__device__ __forceinline__ f32x8 wmma_bf16(bf16x16 a, bf16x16 b, f32x8 c) {
  return __builtin_amdgcn_wmma_f32_16x16x32_bf16(false, a, false, b,
                                                 (short)0, c, false, false);
}

// ---------------------------------------------------------------------------
// Kernel 0: weight conversion. wcat = [theta(64); phi(64); g(256)] x 512 bf16,
// wob = w_o [512][256] bf16.
// ---------------------------------------------------------------------------
__global__ void k_prep_weights(const float* __restrict__ wt,
                               const float* __restrict__ wp,
                               const float* __restrict__ wg,
                               const float* __restrict__ wo,
                               __bf16* __restrict__ wcat,
                               __bf16* __restrict__ wob) {
  int i = blockIdx.x * blockDim.x + threadIdx.x;
  if (i < 384 * 512) {
    int o = i >> 9, c = i & 511;
    float v = (o < 64) ? wt[o * 512 + c]
            : (o < 128) ? wp[(o - 64) * 512 + c]
                        : wg[(o - 128) * 512 + c];
    wcat[i] = f2bf(v);
  }
  if (i < 512 * 256) wob[i] = f2bf(wo[i]);
}

// ---------------------------------------------------------------------------
// Kernel 1: fused projection GEMM + 2x2 maxpool.
// P[o,s] = sum_c wcat[o,c] * x[n,c,s].  One block covers ALL 384 output
// channels for a 128-wide spatial span (= image rows h,h+1, h even):
// 4 waves x 2 N-tiles x 24 o-tiles, 48 f32x8 accumulators per wave
// (384 VGPRs; CDNA5 allows 1024/wave).  x is streamed exactly once and each
// B-fragment build feeds 48 WMMAs.
// theta rows -> thetaT[n][s][64] (transposed, packed 16B stores)
// phi rows   -> pooled -> phiT[n][t][64]   (t-major for A-frags of scores^T)
// g rows     -> pooled -> gpool[n][256][1024] (d-major for A-frags of y GEMM)
// ---------------------------------------------------------------------------
__global__ void __launch_bounds__(128)
k_proj(const float* __restrict__ x, const __bf16* __restrict__ wcat,
       __bf16* __restrict__ thetaT, __bf16* __restrict__ phiT,
       __bf16* __restrict__ gpool) {
  const int lane = threadIdx.x & 31;
  const int wv   = threadIdx.x >> 5;     // 0..3
  const int sblk = blockIdx.x;           // 0..31
  const int n    = blockIdx.y;           // 0..15
  const int l15  = lane & 15, up = lane >> 4;

  const int s0 = sblk * 128 + wv * 16;   // in image row h = 2*sblk
  const int s1 = s0 + 64;                // in image row h+1
  const float* xn = x + (size_t)n * 512 * 4096;

  f32x8 acc0[24], acc1[24];
#pragma unroll
  for (int ot = 0; ot < 24; ++ot) { acc0[ot] = (f32x8){}; acc1[ot] = (f32x8){}; }

  for (int kc = 0; kc < 512; kc += 32) {
    bf16x16 b0, b1;
#pragma unroll
    for (int h = 0; h < 16; ++h) {
      int k = kc + h + ((h >= 8) ? 8 : 0) + (up ? 8 : 0);
      b0[h] = f2bf(xn[(size_t)k * 4096 + s0 + l15]);
      b1[h] = f2bf(xn[(size_t)k * 4096 + s1 + l15]);
    }
#pragma unroll
    for (int ot = 0; ot < 24; ++ot) {
      bf16x16 a = load_frag(wcat + (size_t)(ot * 16) * 512 + kc, 512, lane);
      acc0[ot] = wmma_bf16(a, b0, acc0[ot]);
      acc1[ot] = wmma_bf16(a, b1, acc1[ot]);
    }
  }

  const int tw = (wv * 16 + l15) >> 1;   // pooled column
  const int t  = sblk * 32 + tw;         // pooled flat index
  const bool active = (lane & 1) == 0;
  __bf16* tb = thetaT + (size_t)n * 4096 * 64;

#pragma unroll
  for (int ot = 0; ot < 24; ++ot) {
    if (ot < 4) {
      // theta (o = 0..63): transposed store, 8 consecutive o per lane -> 16B.
      bf16x8 pk0, pk1;
#pragma unroll
      for (int r = 0; r < 8; ++r) { pk0[r] = f2bf(acc0[ot][r]); pk1[r] = f2bf(acc1[ot][r]); }
      const int ob = ot * 16 + up * 8;
      *(bf16x8*)(tb + (size_t)(s0 + l15) * 64 + ob) = pk0;
      *(bf16x8*)(tb + (size_t)(s1 + l15) * 64 + ob) = pk1;
    } else {
      // 2x2 maxpool: vertical pair = acc0/acc1, horizontal pair = lane ^ 1.
#pragma unroll
      for (int r = 0; r < 8; ++r) {
        float v = fmaxf(acc0[ot][r], acc1[ot][r]);
        v = fmaxf(v, __shfl_xor(v, 1, 32));
        if (active) {
          int o = ot * 16 + r + up * 8;
          if (ot < 8) {   // phi: d = o-64, phiT[n][t][d]
            phiT[((size_t)n * 1024 + t) * 64 + (o - 64)] = f2bf(v);
          } else {        // g: dg = o-128, gpool[n][dg][t]
            gpool[((size_t)n * 256 + (o - 128)) * 1024 + t] = f2bf(v);
          }
        }
      }
    }
  }
}

// ---------------------------------------------------------------------------
// Kernel 3: flash attention.  One wave = 16 queries x 1024 keys x 256 dhat.
// Compute S^T (M=key, N=query) so each lane owns one query's scores:
// softmax stats are in-lane + one shfl_xor(16); a 32-key chunk's two S^T
// D-frags map per-lane onto the bf16 B-frag of the P x g GEMM.
// Y (M=dhat, N=query) kept in 16 f32 frags (128 VGPRs); stored as
// yT[n][s][256] with packed 16B stores.
// ---------------------------------------------------------------------------
__global__ void __launch_bounds__(256)
k_attn(const __bf16* __restrict__ thetaT, const __bf16* __restrict__ phiT,
       const __bf16* __restrict__ gpool, __bf16* __restrict__ yT) {
  const int lane = threadIdx.x & 31;
  const int wid  = blockIdx.x * 8 + (threadIdx.x >> 5);
  const int n    = wid >> 8;            // 256 query tiles per batch
  const int qb   = (wid & 255) * 16;
  const int l15  = lane & 15;

  const __bf16* thn = thetaT + (size_t)n * 4096 * 64;
  const __bf16* phn = phiT   + (size_t)n * 1024 * 64;
  const __bf16* gn  = gpool  + (size_t)n * 256 * 1024;

  // B = theta (K=d 0..63, N=query): loaded once, reused for all 64 key tiles.
  bf16x16 bt0 = load_frag(thn + (size_t)qb * 64,      64, lane);
  bf16x16 bt1 = load_frag(thn + (size_t)qb * 64 + 32, 64, lane);

  f32x8 Y[16];
#pragma unroll
  for (int i = 0; i < 16; ++i) Y[i] = (f32x8){};

  float m_run = -3.0e38f, l_run = 0.0f;

  for (int tc = 0; tc < 32; ++tc) {               // 32 chunks of 32 keys
    const __bf16* pt0 = phn + (size_t)(tc * 32) * 64;
    const __bf16* pt1 = pt0 + 16 * 64;
    f32x8 st0 = {}, st1 = {};
    st0 = wmma_bf16(load_frag(pt0,      64, lane), bt0, st0);
    st0 = wmma_bf16(load_frag(pt0 + 32, 64, lane), bt1, st0);
    st1 = wmma_bf16(load_frag(pt1,      64, lane), bt0, st1);
    st1 = wmma_bf16(load_frag(pt1 + 32, 64, lane), bt1, st1);

    float mc = st0[0];
#pragma unroll
    for (int i = 0; i < 8; ++i) { mc = fmaxf(mc, st0[i]); mc = fmaxf(mc, st1[i]); }
    mc = fmaxf(mc, __shfl_xor(mc, 16, 32));       // pair lane holds other keys
    const float mnew = fmaxf(m_run, mc);
    const float corr = __expf(m_run - mnew);

    float ps = 0.0f;
    bf16x16 pf;                                   // P as B-frag (K=key chunk)
#pragma unroll
    for (int i = 0; i < 8; ++i) {
      float e0 = __expf(st0[i] - mnew);
      float e1 = __expf(st1[i] - mnew);
      ps += e0 + e1;
      pf[i]     = f2bf(e0);                       // keys 0..7  (+8 upper)
      pf[i + 8] = f2bf(e1);                       // keys 16..23(+8 upper)
    }
    ps += __shfl_xor(ps, 16, 32);
    l_run = l_run * corr + ps;
    m_run = mnew;

#pragma unroll
    for (int dt = 0; dt < 16; ++dt) {
#pragma unroll
      for (int i = 0; i < 8; ++i) Y[dt][i] *= corr;
      bf16x16 ag = load_frag(gn + (size_t)(dt * 16) * 1024 + tc * 32, 1024, lane);
      Y[dt] = wmma_bf16(ag, pf, Y[dt]);           // Y[dhat, query] += g * P
    }
  }

  const float inv = 1.0f / l_run;
  __bf16* yn = yT + ((size_t)n * 4096 + qb) * 256;
#pragma unroll
  for (int dt = 0; dt < 16; ++dt) {
    bf16x8 pk;
#pragma unroll
    for (int r = 0; r < 8; ++r) pk[r] = f2bf(Y[dt][r] * inv);
    *(bf16x8*)(yn + (size_t)l15 * 256 + dt * 16 + ((lane >> 4) << 3)) = pk;
  }
}

// ---------------------------------------------------------------------------
// Kernel 4: o = w_o x y ; out = gamma*o + x.  M=c so stores are coalesced.
// One wave: 16 c x 64 s, K = 256.
// ---------------------------------------------------------------------------
__global__ void __launch_bounds__(256)
k_out(const float* __restrict__ x, const __bf16* __restrict__ wob,
      const __bf16* __restrict__ yT, const float* __restrict__ gamma,
      float* __restrict__ out) {
  const int lane = threadIdx.x & 31;
  const int wid  = blockIdx.x * 8 + (threadIdx.x >> 5);
  const int sg   = wid & 63;            // spatial group of 64
  const int ct   = (wid >> 6) & 31;     // c tile
  const int n    = wid >> 11;
  const int l15  = lane & 15, up = lane >> 4;

  const __bf16* yn = yT + (size_t)n * 4096 * 256;
  bf16x16 A[8];
#pragma unroll
  for (int k = 0; k < 8; ++k)
    A[k] = load_frag(wob + (size_t)(ct * 16) * 256 + k * 32, 256, lane);

  f32x8 acc[4];
#pragma unroll
  for (int j = 0; j < 4; ++j) acc[j] = (f32x8){};

#pragma unroll
  for (int k = 0; k < 8; ++k) {
#pragma unroll
    for (int j = 0; j < 4; ++j) {
      bf16x16 b = load_frag(yn + (size_t)(sg * 64 + j * 16) * 256 + k * 32, 256, lane);
      acc[j] = wmma_bf16(A[k], b, acc[j]);
    }
  }

  const float g = gamma[0];
  const float* xn = x + (size_t)n * 512 * 4096;
  float* on = out + (size_t)n * 512 * 4096;
#pragma unroll
  for (int j = 0; j < 4; ++j) {
    int s = sg * 64 + j * 16 + l15;
#pragma unroll
    for (int r = 0; r < 8; ++r) {
      size_t idx = (size_t)(ct * 16 + r + up * 8) * 4096 + s;
      on[idx] = g * acc[j][r] + xn[idx];
    }
  }
}

// ---------------------------------------------------------------------------
// Workspace layout (bytes, all 256-aligned):
//   wcat   [384x512 bf16]        @ 0         (393216)
//   wob    [512x256 bf16]        @ 393216    (262144)
//   thetaT [16][4096][64] bf16   @ 655360    (8388608)
//   phiT   [16][1024][64] bf16   @ 9043968   (2097152)
//   gpool  [16][256][1024] bf16  @ 11141120  (8388608)
//   yT     [16][4096][256] bf16  @ 19529728  (33554432)   total ~53 MB
// ---------------------------------------------------------------------------
extern "C" void kernel_launch(void* const* d_in, const int* in_sizes, int n_in,
                              void* d_out, int out_size, void* d_ws, size_t ws_size,
                              hipStream_t stream) {
  const float* x  = (const float*)d_in[0];
  const float* wt = (const float*)d_in[1];
  const float* wp = (const float*)d_in[2];
  const float* wg = (const float*)d_in[3];
  const float* wo = (const float*)d_in[4];
  const float* gm = (const float*)d_in[5];
  float* out = (float*)d_out;

  char* ws = (char*)d_ws;
  __bf16* wcat   = (__bf16*)(ws + 0);
  __bf16* wob    = (__bf16*)(ws + 393216);
  __bf16* thetaT = (__bf16*)(ws + 655360);
  __bf16* phiT   = (__bf16*)(ws + 9043968);
  __bf16* gpool  = (__bf16*)(ws + 11141120);
  __bf16* yT     = (__bf16*)(ws + 19529728);

  k_prep_weights<<<768, 256, 0, stream>>>(wt, wp, wg, wo, wcat, wob);
  k_proj<<<dim3(32, 16), 128, 0, stream>>>(x, wcat, thetaT, phiT, gpool);
  k_attn<<<512, 256, 0, stream>>>(thetaT, phiT, gpool, yT);
  k_out<<<4096, 256, 0, stream>>>(x, wob, yT, gm, out);
}
